// Dgcnn_88012469830112
// MI455X (gfx1250) — compile-verified, matrix operand order checked by reading, not run on hardware
//
#include <hip/hip_runtime.h>

typedef __attribute__((ext_vector_type(16))) _Float16 v16h;
typedef __attribute__((ext_vector_type(8)))  _Float16 v8h;
typedef __attribute__((ext_vector_type(8)))  float    v8f;

#define FLT_BIG 3.0e38f
#define NPTS 4096
#define HID  192
#define CO   384
#define KNN  16
#define TPB  256

// Dynamic LDS layout:
//   [0, 49152)              ptsL    float[NPTS*3]                (48 KB)
//   [49152, 49152+98304)    hL      _Float16[TPB*HID]            (96 KB)
//       (overlapped early): knnD float[TPB*KNN] @ +0, knnI int[TPB*KNN] @ +16384
//   [147456, 148480)        nbrIdx  int[16*KNN]                  (1 KB)
#define SMEM_BYTES (49152 + 98304 + 1024)

#if defined(__has_builtin)
#if __has_builtin(__builtin_amdgcn_global_load_async_to_lds_b128)
#define HAVE_ASYNC_LDS 1
#endif
#if __has_builtin(__builtin_amdgcn_s_wait_asynccnt)
#define HAVE_WAIT_ASYNC 1
#endif
#endif

#if defined(HAVE_ASYNC_LDS)
typedef int v4i_vs __attribute__((vector_size(16)));
typedef __attribute__((address_space(1))) v4i_vs gv4i;   // global int4*
typedef __attribute__((address_space(3))) v4i_vs lv4i;   // LDS int4*
#endif

__global__ __launch_bounds__(TPB) void dgcnn_knn_mlp_wmma(
    const float* __restrict__ point,
    const float* __restrict__ W1,
    const float* __restrict__ b1,
    const float* __restrict__ W2,
    const float* __restrict__ b2,
    float* __restrict__ out)
{
  extern __shared__ char smem[];
  float*    ptsL   = (float*)smem;
  char*     dynr   = smem + NPTS*3*sizeof(float);
  _Float16* hL     = (_Float16*)dynr;
  float*    knnD   = (float*)dynr;
  int*      knnI   = (int*)(dynr + TPB*KNN*sizeof(float));
  int*      nbrIdx = (int*)(dynr + (size_t)TPB*HID*sizeof(_Float16));

  const int tid  = threadIdx.x;
  const int wave = tid >> 5;
  const int lane = tid & 31;
  const int lp   = lane & 15;
  const int half = lane >> 4;

  const int b    = blockIdx.x >> 8;     // 256 tiles per batch
  const int tile = blockIdx.x & 255;
  const int n0   = tile * 16;

  const float* ptsG = point + (size_t)b * NPTS * 3;

  // Warm GL2 for the Phase-4 uniform (scalar) weight reads.
  __builtin_prefetch(W1 + tid, 0, 0);

  // ---- Phase 0: stage this batch's point cloud into LDS ----
#if defined(HAVE_ASYNC_LDS)
  {
    gv4i* g = (gv4i*)(void*)(const void*)ptsG;
    lv4i* l = (lv4i*)(void*)ptsL;
    #pragma unroll
    for (int i = 0; i < (NPTS*3/4)/TPB; ++i)   // 12 x 16B chunks per thread
      __builtin_amdgcn_global_load_async_to_lds_b128(
          g + (tid + i*TPB), l + (tid + i*TPB), 0, 0);
  }
#else
  {
    const float4* src = (const float4*)ptsG;
    float4* dst = (float4*)ptsL;
    #pragma unroll
    for (int i = 0; i < (NPTS*3/4)/TPB; ++i)
      dst[tid + i*TPB] = src[tid + i*TPB];
  }
#endif

  // ---- Phase 1: this wave's W2 B-fragments -> registers (f32 -> f16) ----
  // Wave w owns output columns [w*48, w*48+48): 3 column tiles x 6 K tiles.
  // 16-bit B layout mirror of A: lane<16 -> col c, K = kb+{0..7} and kb+16+{0..7};
  // lane>=16 -> same col set, K offset +8.
  v16h bfrag[3][6];
  #pragma unroll
  for (int ct = 0; ct < 3; ++ct) {
    const int c = wave*48 + ct*16 + lp;
    #pragma unroll
    for (int kt = 0; kt < 6; ++kt) {
      const int kb = kt*32 + half*8;
      v16h bf;
      #pragma unroll
      for (int i = 0; i < 8; ++i) {
        bf[i]   = (_Float16)W2[(size_t)(kb + i     )*CO + c];
        bf[i+8] = (_Float16)W2[(size_t)(kb + 16 + i)*CO + c];
      }
      bfrag[ct][kt] = bf;
    }
  }

#if defined(HAVE_ASYNC_LDS)
  // Async-engine LDS writes are tracked by ASYNCcnt, not by the barrier.
#if defined(HAVE_WAIT_ASYNC)
  __builtin_amdgcn_s_wait_asynccnt(0);
#else
  asm volatile("s_wait_asynccnt 0x0" ::: "memory");
#endif
#endif
  __syncthreads();

  // ---- Phase 2: KNN scan. thread = (query p, stripe s), 256 candidates each ----
  const int p  = tid >> 4;
  const int s  = tid & 15;
  const int nq = n0 + p;
  const float qx = ptsL[nq*3+0], qy = ptsL[nq*3+1], qz = ptsL[nq*3+2];

  float dbest[KNN]; int ibest[KNN];
  #pragma unroll
  for (int i = 0; i < KNN; ++i) { dbest[i] = FLT_BIG; ibest[i] = 0; }
  float dmax = FLT_BIG; int amax = 0;

  const int j0 = s * (NPTS/16);
  for (int jj = j0; jj < j0 + NPTS/16; ++jj) {
    const float dx = ptsL[jj*3+0] - qx;
    const float dy = ptsL[jj*3+1] - qy;
    const float dz = ptsL[jj*3+2] - qz;
    const float dd = dx*dx + dy*dy + dz*dz;
    if (dd < dmax) {                       // rare accept: replace current max slot
      #pragma unroll
      for (int t2 = 0; t2 < KNN; ++t2)
        if (t2 == amax) { dbest[t2] = dd; ibest[t2] = jj; }
      dmax = dbest[0]; amax = 0;
      #pragma unroll
      for (int t2 = 1; t2 < KNN; ++t2)
        if (dbest[t2] > dmax) { dmax = dbest[t2]; amax = t2; }
    }
  }
  #pragma unroll
  for (int i = 0; i < KNN; ++i) {
    knnD[tid*KNN + i] = dbest[i];
    knnI[tid*KNN + i] = ibest[i];
  }
  __syncthreads();

  // ---- Phase 3: merge 16 partial top-16 lists per query (leader lanes) ----
  if (s == 0) {
    float md[KNN]; int mi[KNN];
    #pragma unroll
    for (int i = 0; i < KNN; ++i) { md[i] = FLT_BIG; mi[i] = 0; }
    float mmax = FLT_BIG; int mam = 0;
    for (int q2 = 0; q2 < 256; ++q2) {
      const float dd = knnD[p*256 + q2];
      const int   ii = knnI[p*256 + q2];
      if (dd < mmax) {
        #pragma unroll
        for (int t2 = 0; t2 < KNN; ++t2)
          if (t2 == mam) { md[t2] = dd; mi[t2] = ii; }
        mmax = md[0]; mam = 0;
        #pragma unroll
        for (int t2 = 1; t2 < KNN; ++t2)
          if (md[t2] > mmax) { mmax = md[t2]; mam = t2; }
      }
    }
    #pragma unroll
    for (int i = 0; i < KNN; ++i) nbrIdx[p*KNN + i] = mi[i];
  }
  __syncthreads();

  // ---- Phase 4: fc1 + swish -> hL (row t = query p, neighbor s), f16 ----
  {
    const int nb = nbrIdx[p*KNN + s];
    const float f0 = ptsL[nb*3+0] - qx;
    const float f1 = ptsL[nb*3+1] - qy;
    const float f2 = ptsL[nb*3+2] - qz;
    unsigned int* hrow = (unsigned int*)(hL + (size_t)tid*HID);
    for (int c = 0; c < HID; c += 2) {   // W1/b1 reads are uniform -> scalar loads
      float a0 = b1[c]   + f0*W1[0*HID+c]   + f1*W1[1*HID+c]   + f2*W1[2*HID+c]
                         + qx*W1[3*HID+c]   + qy*W1[4*HID+c]   + qz*W1[5*HID+c];
      float a1 = b1[c+1] + f0*W1[0*HID+c+1] + f1*W1[1*HID+c+1] + f2*W1[2*HID+c+1]
                         + qx*W1[3*HID+c+1] + qy*W1[4*HID+c+1] + qz*W1[5*HID+c+1];
      a0 = a0 / (1.0f + __expf(-a0));    // x*sigmoid(x)
      a1 = a1 / (1.0f + __expf(-a1));
      union { _Float16 h[2]; unsigned int u; } pk;
      pk.h[0] = (_Float16)a0; pk.h[1] = (_Float16)a1;
      hrow[c >> 1] = pk.u;
    }
  }
  __syncthreads();

  // ---- Phase 5: fc2 via v_wmma_f32_16x16x32_f16, fused row-max + bias + store ----
  const float bias0 = b2[wave*48 +  0 + lp];
  const float bias1 = b2[wave*48 + 16 + lp];
  const float bias2 = b2[wave*48 + 32 + lp];
  const size_t outrow = ((size_t)b * NPTS + n0) * CO;

  for (int m = 0; m < 16; ++m) {         // M-tile m = query point m's 16 neighbor rows
    v8f acc0 = {}, acc1 = {}, acc2 = {};
    #pragma unroll
    for (int kt = 0; kt < 6; ++kt) {
      // A fragment: row = m*16+lp, K = kt*32 + half*8 + {0..7} and +16
      const _Float16* ab = hL + (size_t)(m*16 + lp)*HID + kt*32 + half*8;
      const v8h lo = *(const v8h*)ab;         // ds_load_b128
      const v8h hi = *(const v8h*)(ab + 16);  // ds_load_b128
      v16h a;
      #pragma unroll
      for (int i = 0; i < 8; ++i) { a[i] = lo[i]; a[i+8] = hi[i]; }
      acc0 = __builtin_amdgcn_wmma_f32_16x16x32_f16(false, a, false, bfrag[0][kt], (short)0, acc0, false, false);
      acc1 = __builtin_amdgcn_wmma_f32_16x16x32_f16(false, a, false, bfrag[1][kt], (short)0, acc1, false, false);
      acc2 = __builtin_amdgcn_wmma_f32_16x16x32_f16(false, a, false, bfrag[2][kt], (short)0, acc2, false, false);
    }
    // max over the 16 rows (k neighbors): 8 VGPRs in-lane, then swap halves (M+8)
    float r0 = acc0[0], r1 = acc1[0], r2 = acc2[0];
    #pragma unroll
    for (int i = 1; i < 8; ++i) {
      r0 = fmaxf(r0, acc0[i]); r1 = fmaxf(r1, acc1[i]); r2 = fmaxf(r2, acc2[i]);
    }
    r0 = fmaxf(r0, __uint_as_float(__builtin_amdgcn_ds_swizzle(__float_as_uint(r0), 0x401F)));
    r1 = fmaxf(r1, __uint_as_float(__builtin_amdgcn_ds_swizzle(__float_as_uint(r1), 0x401F)));
    r2 = fmaxf(r2, __uint_as_float(__builtin_amdgcn_ds_swizzle(__float_as_uint(r2), 0x401F)));
    if (half == 0) {
      float* o = out + outrow + (size_t)m*CO + wave*48 + lp;
      o[0]  = r0 + bias0;
      o[16] = r1 + bias1;
      o[32] = r2 + bias2;
    }
  }
}

extern "C" void kernel_launch(void* const* d_in, const int* in_sizes, int n_in,
                              void* d_out, int out_size, void* d_ws, size_t ws_size,
                              hipStream_t stream) {
  (void)n_in; (void)out_size; (void)d_ws; (void)ws_size;
  const float* point = (const float*)d_in[0];
  const float* W1    = (const float*)d_in[1];
  const float* b1    = (const float*)d_in[2];
  const float* W2    = (const float*)d_in[3];
  const float* b2    = (const float*)d_in[4];
  float* out = (float*)d_out;

  const int B = in_sizes[0] / (NPTS * 3);

  (void)hipFuncSetAttribute(reinterpret_cast<const void*>(&dgcnn_knn_mlp_wmma),
                            hipFuncAttributeMaxDynamicSharedMemorySize, SMEM_BYTES);

  dgcnn_knn_mlp_wmma<<<B * (NPTS/16), TPB, SMEM_BYTES, stream>>>(
      point, W1, b1, W2, b2, out);
}
